// RNN_16836271800341
// MI455X (gfx1250) — compile-verified, matrix operand order checked by reading, not run on hardware
//
#include <hip/hip_runtime.h>
#include <hip/hip_bf16.h>
#include <math.h>

// Problem dims
#define B_SZ  256
#define T_SZ  256
#define NIN   128
#define NHID  1024
#define NRES  512
#define NOUT  64

// WMMA vector types (gfx1250, wave32)
typedef __attribute__((ext_vector_type(16))) __bf16 v16bf;
typedef __attribute__((ext_vector_type(8)))  float  v8f;

union ABu { int4 q[2]; v16bf v; };

// Packed bf16 weight layout offsets (in bf16 elements).
// Each 16x32 B-tile = 512 bf16 (1 KB); within a tile, lane L owns 32 contiguous bytes.
#define OFF_WIN   0
#define OFF_WHH   131072
#define OFF_WSF   1179648
#define OFF_WFS   1703936
#define OFF_WRES  2228224
#define OFF_WOUT  2490368
#define TOT_W     2555904
#define BIAS_BYTE_OFF (TOT_W * 2)      // f32 combined biases follow packed weights

__device__ __forceinline__ unsigned short f2bf(float f) {
  union { float f; unsigned int u; } x; x.f = f;
  unsigned int u = x.u;
  u += 0x7FFFu + ((u >> 16) & 1u);     // round-to-nearest-even
  return (unsigned short)(u >> 16);
}

__device__ __forceinline__ float bf2f(unsigned short h) {
  union { unsigned int u; float f; } x; x.u = ((unsigned int)h) << 16; return x.f;
}

__device__ __forceinline__ v8f splat8(float v) {
  v8f a;
#pragma unroll
  for (int e = 0; e < 8; ++e) a[e] = v;
  return a;
}

// Deterministic counter-based pair of Gaussians (hash + Box-Muller, 2 outputs).
__device__ __forceinline__ float2 gauss2(unsigned int key) {
  unsigned int s = key * 0x9E3779B9u;
  s ^= s >> 16; s *= 0x7FEB352Du; s ^= s >> 15; s *= 0x846CA68Bu; s ^= s >> 16;
  unsigned int q = s * 0x41C64E6Du + 12345u;
  float u1 = (float)((s >> 8) | 1u) * 5.9604645e-8f;   // (0,1)
  float u2 = (float)(q >> 8) * 5.9604645e-8f;
  float rr = sqrtf(-2.0f * __logf(u1));
  float sn, cs;
  __sincosf(6.2831853f * u2, &sn, &cs);
  float2 o; o.x = rr * cs; o.y = rr * sn;
  return o;
}

// Register-blocked WMMA: NT 16x16 tiles share each A-fragment.
// aLds: bf16 activations [16 x aStride] in LDS.
// wSeg: packed weights, tile (jt0, kt=0); tile (jt0+i, kt=c) at + i*tileStride + c*512.
template <int NT>
__device__ __forceinline__ void gemm_block(v8f* acc,
                                           const unsigned short* aLds, int aStride,
                                           int nChunks,
                                           const unsigned short* wSeg, int tileStride,
                                           int lane) {
  const int m   = lane & 15;
  const int khi = (lane & 16) ? 8 : 0;
  const unsigned short* wp = wSeg + lane * 16;   // 32 bytes per lane
#pragma unroll 2
  for (int c = 0; c < nChunks; ++c) {
    ABu a;
    const unsigned short* ap = aLds + m * aStride + (c << 5) + khi;
    a.q[0] = *(const int4*)(ap);        // K = c*32+khi    .. +7  -> A VGPR 0..3
    a.q[1] = *(const int4*)(ap + 16);   // K = c*32+khi+16 .. +23 -> A VGPR 4..7
#pragma unroll
    for (int i = 0; i < NT; ++i) {
      ABu b;
      const unsigned short* bp = wp + i * tileStride;
      b.q[0] = *(const int4*)(bp);
      b.q[1] = *(const int4*)(bp + 8);
      acc[i] = __builtin_amdgcn_wmma_f32_16x16x32_bf16(false, a.v, false, b.v,
                                                       (short)0, acc[i], false, false);
    }
    wp += 512;
  }
}

// ---------------- Pre-pack: f32 weights -> bf16 WMMA-B-tiled layout ----------------
__global__ void pack_weights(const float* __restrict__ win, const float* __restrict__ whh,
                             const float* __restrict__ wsf, const float* __restrict__ wfs,
                             const float* __restrict__ wres, const float* __restrict__ wout,
                             unsigned int* __restrict__ dst) {
  int p = blockIdx.x * blockDim.x + threadIdx.x;    // one packed dword (bf16 pair)
  if (p >= TOT_W / 2) return;
  const float* src; int q, nin, numK;
  if (p < OFF_WHH / 2)       { src = win;  q = p;                nin = NIN;  numK = NIN / 32; }
  else if (p < OFF_WSF / 2)  { src = whh;  q = p - OFF_WHH / 2;  nin = NHID; numK = 32; }
  else if (p < OFF_WFS / 2)  { src = wsf;  q = p - OFF_WSF / 2;  nin = NRES; numK = 16; }
  else if (p < OFF_WRES / 2) { src = wfs;  q = p - OFF_WFS / 2;  nin = NHID; numK = 32; }
  else if (p < OFF_WOUT / 2) { src = wres; q = p - OFF_WRES / 2; nin = NRES; numK = 16; }
  else                       { src = wout; q = p - OFF_WOUT / 2; nin = NHID; numK = 32; }
  int tile = q >> 8;          // 256 dwords per tile
  int d    = q & 255;
  int lane = d >> 3, v = d & 7;
  int jt = tile / numK, kt = tile - jt * numK;
  int j = jt * 16 + (lane & 15);
  int k = kt * 32 + ((v & 4) ? 16 : 0) + ((lane & 16) ? 8 : 0) + ((v & 3) << 1);
  unsigned lo = f2bf(src[(size_t)j * nin + k]);
  unsigned hi = f2bf(src[(size_t)j * nin + k + 1]);
  dst[p] = lo | (hi << 16);
}

__global__ void combine_bias(const float* __restrict__ bin, const float* __restrict__ bhh,
                             const float* __restrict__ bsf, const float* __restrict__ bfs,
                             const float* __restrict__ bres, const float* __restrict__ bout,
                             float* __restrict__ bias) {
  int j = blockIdx.x * blockDim.x + threadIdx.x;
  if (j < NHID)                    bias[j] = bin[j] + bhh[j] + bsf[j];
  else if (j < NHID + NRES)        bias[j] = bfs[j - NHID] + bres[j - NHID];
  else if (j < NHID + NRES + NOUT) bias[j] = bout[j - NHID - NRES];
}

// ---------------- Persistent RNN scan: 1 workgroup = 16 batch rows, all T steps ----
// LDS state is double-buffered so each tile commits immediately after its GEMM
// (no long-lived accumulator arrays -> no spills), with 2 barriers per step.
__global__ __launch_bounds__(512) void rnn_persistent(
    const float* __restrict__ x, const float* __restrict__ h0, const float* __restrict__ r0,
    const unsigned short* __restrict__ wpk, const float* __restrict__ bias,
    float* __restrict__ oh, float* __restrict__ oo, float* __restrict__ orr) {
  __shared__ __align__(16) unsigned short h_buf[2][16 * NHID];  // 2 x 32 KB
  __shared__ __align__(16) unsigned short r_buf[2][16 * NRES];  // 2 x 16 KB
  __shared__ __align__(16) unsigned short x_bf[16 * NIN];       //     4 KB

  const int tid  = threadIdx.x;
  const int lane = tid & 31;
  const int wave = tid >> 5;                // 16 waves
  const int row0 = blockIdx.x * 16;         // batch tile base
  const int m    = lane & 15;
  const int khi  = (lane & 16) ? 8 : 0;

  // Load initial state into buffer 0.
  for (int i = tid; i < 16 * NHID; i += 512)
    h_buf[0][i] = f2bf(h0[(size_t)(row0 + (i >> 10)) * NHID + (i & (NHID - 1))]);
  for (int i = tid; i < 16 * NRES; i += 512)
    r_buf[0][i] = f2bf(r0[(size_t)(row0 + (i >> 9)) * NRES + (i & (NRES - 1))]);
  __syncthreads();

  unsigned cur = 0;
  for (int t = 0; t < T_SZ; ++t) {
    const unsigned short* hC = h_buf[cur];
    const unsigned short* rC = r_buf[cur];
    unsigned short*       hW = h_buf[cur ^ 1];
    unsigned short*       rW = r_buf[cur ^ 1];

    // Stage x_t tile (16 x 128) as bf16.
    for (int i = tid; i < 16 * NIN; i += 512) {
      int mr = i >> 7, c = i & (NIN - 1);
      x_bf[i] = f2bf(x[((size_t)(row0 + mr) * T_SZ + t) * NIN + c]);
    }
    __syncthreads();                         // barrier A: x staged, state stable

    // ---- hidden: this wave's 4 contiguous j-tiles, register-blocked.
    {
      v8f acc[4];
#pragma unroll
      for (int i = 0; i < 4; ++i) acc[i] = splat8(bias[(wave * 4 + i) * 16 + m]);
      gemm_block<4>(acc, x_bf, NIN, NIN / 32,
                    wpk + OFF_WIN + (size_t)(wave * 4) * (NIN / 32) * 512, (NIN / 32) * 512, lane);
      gemm_block<4>(acc, hC, NHID, 32,
                    wpk + OFF_WHH + (size_t)(wave * 4) * 32 * 512, 32 * 512, lane);
      gemm_block<4>(acc, rC, NRES, 16,
                    wpk + OFF_WSF + (size_t)(wave * 4) * 16 * 512, 16 * 512, lane);
#pragma unroll
      for (int i = 0; i < 4; ++i) {
        int j = (wave * 4 + i) * 16 + m;
#pragma unroll
        for (int e = 0; e < 8; e += 2) {
          int mr = e + khi;
          float v0 = fmaxf(acc[i][e], 0.0f);
          float v1 = fmaxf(acc[i][e + 1], 0.0f);
          unsigned key = (((unsigned)t * 256u + (unsigned)(row0 + mr)) << 11) + (unsigned)j;
          float2 g = gauss2(key);
          v0 += 0.05f * g.x;                 // alpha_fast = 1
          v1 += 0.05f * g.y;
          hW[mr * NHID + j]       = f2bf(v0);
          hW[(mr + 1) * NHID + j] = f2bf(v1);
          oh[((size_t)(row0 + mr) * T_SZ + t) * NHID + j]     = v0;
          oh[((size_t)(row0 + mr + 1) * T_SZ + t) * NHID + j] = v1;
        }
      }
    }
    // ---- reservoir: this wave's 2 contiguous j-tiles.
    {
      v8f acc[2];
#pragma unroll
      for (int i = 0; i < 2; ++i) acc[i] = splat8(bias[NHID + (wave * 2 + i) * 16 + m]);
      gemm_block<2>(acc, hC, NHID, 32,
                    wpk + OFF_WFS + (size_t)(wave * 2) * 32 * 512, 32 * 512, lane);
      gemm_block<2>(acc, rC, NRES, 16,
                    wpk + OFF_WRES + (size_t)(wave * 2) * 16 * 512, 16 * 512, lane);
#pragma unroll
      for (int i = 0; i < 2; ++i) {
        int j = (wave * 2 + i) * 16 + m;
#pragma unroll
        for (int e = 0; e < 8; e += 2) {
          int mr = e + khi;
          float t0 = fmaxf(acc[i][e], 0.0f);
          float t1 = fmaxf(acc[i][e + 1], 0.0f);
          float o0 = bf2f(rC[mr * NRES + j]);
          float o1 = bf2f(rC[(mr + 1) * NRES + j]);
          unsigned key = (((unsigned)t * 256u + (unsigned)(row0 + mr)) << 11) + 1024u + (unsigned)j;
          float2 g = gauss2(key);
          float v0 = 0.9f * o0 + 0.1f * t0 + 0.015811388f * g.x;
          float v1 = 0.9f * o1 + 0.1f * t1 + 0.015811388f * g.y;
          rW[mr * NRES + j]       = f2bf(v0);
          rW[(mr + 1) * NRES + j] = f2bf(v1);
          orr[((size_t)(row0 + mr) * T_SZ + t) * NRES + j]     = v0;
          orr[((size_t)(row0 + mr + 1) * T_SZ + t) * NRES + j] = v1;
        }
      }
    }
    __syncthreads();                         // barrier B: new state committed

    // ---- output head from UPDATED h: 4 tiles on waves 0..3 (wave-uniform branch)
    if (wave < 4) {
      v8f acc[1];
      acc[0] = splat8(bias[NHID + NRES + wave * 16 + m]);
      gemm_block<1>(acc, hW, NHID, 32,
                    wpk + OFF_WOUT + (size_t)wave * 32 * 512, 32 * 512, lane);
#pragma unroll
      for (int e = 0; e < 8; ++e) {
        float v = fminf(2.0f, fmaxf(-2.0f, acc[0][e]));
        oo[((size_t)(row0 + e + khi) * T_SZ + t) * NOUT + wave * 16 + m] = v;
      }
    }
    // Next iteration writes h_buf[cur] (the buffer the out-head is NOT reading),
    // and only after barrier A of t+1 which waves 0-3 reach after the head GEMM.
    cur ^= 1;
  }
}

extern "C" void kernel_launch(void* const* d_in, const int* in_sizes, int n_in,
                              void* d_out, int out_size, void* d_ws, size_t ws_size,
                              hipStream_t stream) {
  (void)in_sizes; (void)n_in; (void)out_size; (void)ws_size;
  const float* x    = (const float*)d_in[0];
  const float* h0   = (const float*)d_in[1];
  const float* r0   = (const float*)d_in[2];
  const float* Win  = (const float*)d_in[3];
  const float* bin  = (const float*)d_in[4];
  const float* Whh  = (const float*)d_in[5];
  const float* bhh  = (const float*)d_in[6];
  const float* Wsf  = (const float*)d_in[7];
  const float* bsf  = (const float*)d_in[8];
  const float* Wfs  = (const float*)d_in[9];
  const float* bfs  = (const float*)d_in[10];
  const float* Wres = (const float*)d_in[11];
  const float* bres = (const float*)d_in[12];
  const float* Wout = (const float*)d_in[13];
  const float* bout = (const float*)d_in[14];

  unsigned short* wpk  = (unsigned short*)d_ws;
  float*          bias = (float*)((char*)d_ws + BIAS_BYTE_OFF);
  float* oh  = (float*)d_out;                                    // [B,T,NHID]
  float* oo  = oh + (size_t)B_SZ * T_SZ * NHID;                  // [B,T,NOUT]
  float* orr = oo + (size_t)B_SZ * T_SZ * NOUT;                  // [B,T,NRES]

  pack_weights<<<(TOT_W / 2 + 255) / 256, 256, 0, stream>>>(
      Win, Whh, Wsf, Wfs, Wres, Wout, (unsigned int*)wpk);
  combine_bias<<<(NHID + NRES + NOUT + 255) / 256, 256, 0, stream>>>(
      bin, bhh, bsf, bfs, bres, bout, bias);
  rnn_persistent<<<B_SZ / 16, 512, 0, stream>>>(x, h0, r0, wpk, bias, oh, oo, orr);
}
